// GroupedGatedDeltaNet_69355131896521
// MI455X (gfx1250) — compile-verified, matrix operand order checked by reading, not run on hardware
//
#include <hip/hip_runtime.h>
#include <hip/hip_bf16.h>

typedef __bf16 bf16;
typedef __attribute__((ext_vector_type(16))) __bf16 v16bf;
typedef __attribute__((ext_vector_type(8)))  float  v8f;

#define B_ 2
#define T_ 1024
#define D_ 2048
#define H_ 16
#define DK_ 128
#define DV_ 128
#define NG_ 8
#define SCALE_ 0.08838834764831845f  // 128^-0.5

union FragBF {
    v16bf   v;
    unsigned u[8];
};

// Packed-pair dword offset inside a [16]-dword K-pair row for fragment dword j.
// Per ISA 7.12.2 (16-bit 16x32 fragments): dword j<4 holds K = 2j,2j+1 (half0)
// or 8+2j.. (half1); dword j>=4 holds K = 16+.. / 24+.. .  In K-pair units the
// bases 0/8/16/24 become 0/4/8/12.
__device__ __forceinline__ int pair_off(int j, int half) {
    return ((j < 4) ? (half ? 4 : 0) : (half ? 12 : 8)) + (j & 3);
}

// ---------------------------------------------------------------------------
// Dense C = A[MxK] @ B[KxN] (f32 in/out, bf16 WMMA math).
// Block 256 thr = 8 waves (4x2). Block tile 128x128, wave tile 32x64,
// 8 accumulators / wave, 8 WMMAs per 32-wide K-step.
// LDS holds the K-tile as packed bf16 pairs: As[row][kpair], Bs[col][kpair].
// Requires M%128==0, N%128==0, K%32==0.
// ---------------------------------------------------------------------------
__global__ __launch_bounds__(256) void wmma_gemm_bf16_fast(
    const float* __restrict__ A, const float* __restrict__ Bm,
    float* __restrict__ C, int M, int N, int K)
{
    __shared__ unsigned As[128 * 16];   // [row 0..127][kpair 0..15]
    __shared__ unsigned Bs[128 * 16];   // [col 0..127][kpair 0..15]
    const int tid  = threadIdx.x;
    const int wave = tid >> 5;
    const int lane = tid & 31;
    const int half = lane >> 4;
    const int l16  = lane & 15;
    const int wm   = wave >> 1;          // 0..3  (M)
    const int wn   = wave & 1;           // 0..1  (N)
    const int m0 = blockIdx.y * 128;
    const int n0 = blockIdx.x * 128;

    v8f acc[2][4];
#pragma unroll
    for (int mi = 0; mi < 2; ++mi)
#pragma unroll
        for (int ni = 0; ni < 4; ++ni) {
            v8f z = {};
            acc[mi][ni] = z;
        }

    for (int kk = 0; kk < K; kk += 32) {
        if (kk + 32 < K) {   // prefetch next K-tile (global_prefetch_b8)
            __builtin_prefetch(&A[(size_t)(m0 + (tid & 127)) * K + kk + 32], 0, 0);
            __builtin_prefetch(&Bm[(size_t)(kk + 32 + (tid >> 3)) * N + n0 + (tid & 7) * 16], 0, 0);
        }
        // stage A 128x32 -> packed bf16 pairs (coalesced: pairs along K)
        for (int i = tid; i < 2048; i += 256) {
            int r = i >> 4, cp = i & 15;
            const float* src = &A[(size_t)(m0 + r) * K + kk + 2 * cp];
            FragBF dummy; (void)dummy;
            union { __bf16 h[2]; unsigned u; } p;
            p.h[0] = (bf16)src[0];
            p.h[1] = (bf16)src[1];
            As[i] = p.u;
        }
        // stage B 32x128 -> [col][kpair] (coalesced: consecutive lanes -> consecutive n)
        for (int i = tid; i < 2048; i += 256) {
            int n = i & 127, kp = i >> 7;
            union { __bf16 h[2]; unsigned u; } p;
            p.h[0] = (bf16)Bm[(size_t)(kk + 2 * kp)     * N + n0 + n];
            p.h[1] = (bf16)Bm[(size_t)(kk + 2 * kp + 1) * N + n0 + n];
            Bs[n * 16 + kp] = p.u;
        }
        __syncthreads();

        FragBF bfr[4];
#pragma unroll
        for (int ni = 0; ni < 4; ++ni) {
            int col = wn * 64 + ni * 16 + l16;
#pragma unroll
            for (int j = 0; j < 8; ++j)
                bfr[ni].u[j] = Bs[col * 16 + pair_off(j, half)];
        }
#pragma unroll
        for (int mi = 0; mi < 2; ++mi) {
            FragBF af;
            int row = wm * 32 + mi * 16 + l16;
#pragma unroll
            for (int j = 0; j < 8; ++j)
                af.u[j] = As[row * 16 + pair_off(j, half)];
#pragma unroll
            for (int ni = 0; ni < 4; ++ni)
                acc[mi][ni] = __builtin_amdgcn_wmma_f32_16x16x32_bf16(
                    false, af.v, false, bfr[ni].v, (short)0, acc[mi][ni], false, false);
        }
        __syncthreads();
    }
    // store: C-frag dword r -> row +r+8*half, col l16
#pragma unroll
    for (int mi = 0; mi < 2; ++mi)
#pragma unroll
        for (int ni = 0; ni < 4; ++ni)
#pragma unroll
            for (int r = 0; r < 8; ++r) {
                int row = m0 + wm * 32 + mi * 16 + r + (half ? 8 : 0);
                int col = n0 + wn * 64 + ni * 16 + l16;
                C[(size_t)row * N + col] = acc[mi][ni][r];
            }
}

// ---------------------------------------------------------------------------
// Generic narrow-N fallback (N%16==0): one 16-col strip per block, 8 waves
// stacked along M.  Used only for the tiny Wb projection (N=16).
// ---------------------------------------------------------------------------
__device__ __forceinline__ int frag_k(int e, int half) {
    int j = e >> 1, p = e & 1;
    int base = (j < 4) ? (half ? 8 : 0) : (half ? 24 : 16);
    return base + ((j & 3) << 1) + p;
}

__global__ __launch_bounds__(256) void wmma_gemm_bf16(
    const float* __restrict__ A, const float* __restrict__ Bm,
    float* __restrict__ C, int M, int N, int K)
{
    __shared__ float As[128 * 32];
    __shared__ float Bs[32 * 16];
    const int tid  = threadIdx.x;
    const int wave = tid >> 5;
    const int lane = tid & 31;
    const int half = lane >> 4;
    const int l16  = lane & 15;
    const int m0 = blockIdx.y * 128;
    const int n0 = blockIdx.x * 16;

    v8f acc = {};
    for (int kk = 0; kk < K; kk += 32) {
        for (int i = tid; i < 128 * 32; i += 256) {
            int r = i >> 5, c = i & 31;
            As[i] = A[(size_t)(m0 + r) * K + (kk + c)];
        }
        for (int i = tid; i < 32 * 16; i += 256) {
            int r = i >> 4, c = i & 15;
            Bs[i] = Bm[(size_t)(kk + r) * N + (n0 + c)];
        }
        __syncthreads();

        v16bf af, bfv;
        const int arow = wave * 16 + l16;
#pragma unroll
        for (int e = 0; e < 16; ++e) {
            int kA = frag_k(e, half);
            af[e]  = (bf16)As[arow * 32 + kA];
            bfv[e] = (bf16)Bs[kA * 16 + l16];
        }
        acc = __builtin_amdgcn_wmma_f32_16x16x32_bf16(
            false, af, false, bfv, (short)0, acc, false, false);
        __syncthreads();
    }
#pragma unroll
    for (int r = 0; r < 8; ++r) {
        int row = m0 + wave * 16 + r + (half ? 8 : 0);
        C[(size_t)row * N + (n0 + l16)] = acc[r];
    }
}

// ---------------------------------------------------------------------------
// Causal depthwise conv1d (K=4) + SiLU, optional per-head L2 norm * scale.
// grid = (B*T, H), block = 128 threads.
// ---------------------------------------------------------------------------
__global__ __launch_bounds__(128) void conv_silu_norm(
    const float* __restrict__ x, const float* __restrict__ w,
    float* __restrict__ y, int T, int do_norm, float scale)
{
    const int bt  = blockIdx.x;          // b*T + t
    const int t   = bt % T;
    const int h   = blockIdx.y;
    const int tid = threadIdx.x;
    const int c   = h * 128 + tid;

    float acc = 0.f;
#pragma unroll
    for (int j = 0; j < 4; ++j) {
        int tp = t - 3 + j;
        if (tp >= 0)
            acc += x[(size_t)(bt - t + tp) * 2048 + c] * w[c * 4 + j];
    }
    float yv = acc / (1.f + expf(-acc));   // silu

    if (do_norm) {
        __shared__ float red[128];
        red[tid] = yv * yv;
        __syncthreads();
        for (int s = 64; s > 0; s >>= 1) {
            if (tid < s) red[tid] += red[tid + s];
            __syncthreads();
        }
        yv *= rsqrtf(red[0] + 1e-6f) * scale;
    }
    y[(size_t)bt * 2048 + c] = yv;
}

// ---------------------------------------------------------------------------
__global__ void gate_kernel(const float* __restrict__ g2,
                            const float* __restrict__ dt_bias,
                            const float* __restrict__ A_log,
                            float* __restrict__ g_eff, int total)
{
    int i = blockIdx.x * blockDim.x + threadIdx.x;
    if (i >= total) return;
    int dk = i & 127;
    int h  = (i >> 7) & 15;
    int bt = i >> 11;
    int ng = dk >> 4;                    // GG = 16
    float xg = g2[(size_t)bt * 128 + h * NG_ + ng] + dt_bias[h * NG_ + ng];
    float sp = (xg > 20.f) ? xg : log1pf(expf(xg));
    g_eff[i] = -expf(A_log[h]) * sp;
}

__global__ void sigmoid_kernel(const float* __restrict__ x,
                               float* __restrict__ y, int total)
{
    int i = blockIdx.x * blockDim.x + threadIdx.x;
    if (i >= total) return;
    y[i] = 1.f / (1.f + expf(-x[i]));
}

// ---------------------------------------------------------------------------
// Gated delta-rule recurrence. One block per (b,h); 256 threads.
// Thread (khalf, vcol) owns S[64*khalf .. +64)[vcol] in VGPRs.
// ---------------------------------------------------------------------------
__global__ __launch_bounds__(256) void recurrence_kernel(
    const float* __restrict__ q, const float* __restrict__ k,
    const float* __restrict__ v, const float* __restrict__ g,
    const float* __restrict__ beta, float* __restrict__ o, int T)
{
    const int bh    = blockIdx.x;
    const int b     = bh >> 4;
    const int h     = bh & 15;
    const int tid   = threadIdx.x;
    const int vcol  = tid & 127;
    const int khalf = tid >> 7;
    const int kbase = khalf * 64;

    __shared__ float q_s[128], k_s[128], eg_s[128], v_s[128];
    __shared__ float err_s[2][128], o_s[2][128];
    __shared__ float beta_s;

    float S[64];
#pragma unroll
    for (int i = 0; i < 64; ++i) S[i] = 0.f;

    for (int t = 0; t < T; ++t) {
        const size_t base = (((size_t)b * T + t) * 16 + h) * 128;
        if (khalf == 0) {
            q_s[vcol]  = q[base + vcol];
            k_s[vcol]  = k[base + vcol];
            eg_s[vcol] = expf(g[base + vcol]);
            v_s[vcol]  = v[base + vcol];
            if (vcol == 0) beta_s = beta[((size_t)b * T + t) * 16 + h];
        }
        __syncthreads();

        float err = 0.f;
#pragma unroll
        for (int i = 0; i < 64; ++i) {
            S[i] *= eg_s[kbase + i];
            err  += k_s[kbase + i] * S[i];
        }
        err_s[khalf][vcol] = err;
        __syncthreads();

        float delta = beta_s * (v_s[vcol] - (err_s[0][vcol] + err_s[1][vcol]));
        float oo = 0.f;
#pragma unroll
        for (int i = 0; i < 64; ++i) {
            S[i] += k_s[kbase + i] * delta;
            oo   += q_s[kbase + i] * S[i];
        }
        o_s[khalf][vcol] = oo;
        __syncthreads();

        if (khalf == 0) o[base + vcol] = o_s[0][vcol] + o_s[1][vcol];
    }
}

// ---------------------------------------------------------------------------
// y = rmsnorm(o * silu(gate_pre + bg)) * norm_w   (per (b,t,h), over 128 dv)
// ---------------------------------------------------------------------------
__global__ __launch_bounds__(128) void out_norm_kernel(
    const float* __restrict__ o, const float* __restrict__ gp,
    const float* __restrict__ bg, const float* __restrict__ norm_w,
    float* __restrict__ y)
{
    const int idx = blockIdx.x;          // (b*T + t)*16 + h
    const int dv  = threadIdx.x;
    const int h   = idx & 15;
    const size_t base = (size_t)idx * 128;

    float gpre = gp[base + dv] + bg[h * 128 + dv];
    float yv   = o[base + dv] * (gpre / (1.f + expf(-gpre)));

    __shared__ float red[128];
    red[dv] = yv * yv;
    __syncthreads();
    for (int s = 64; s > 0; s >>= 1) {
        if (dv < s) red[dv] += red[dv + s];
        __syncthreads();
    }
    float inv = rsqrtf(red[0] * (1.f / 128.f) + 1e-5f);
    y[base + dv] = yv * inv * norm_w[dv];
}

// ---------------------------------------------------------------------------
extern "C" void kernel_launch(void* const* d_in, const int* in_sizes, int n_in,
                              void* d_out, int out_size, void* d_ws, size_t ws_size,
                              hipStream_t stream) {
    (void)in_sizes; (void)n_in; (void)out_size; (void)ws_size;
    const float* hidden  = (const float*)d_in[0];
    const float* Wq      = (const float*)d_in[1];
    const float* Wk      = (const float*)d_in[2];
    const float* Wv      = (const float*)d_in[3];
    const float* conv_q  = (const float*)d_in[4];
    const float* conv_k  = (const float*)d_in[5];
    const float* conv_v  = (const float*)d_in[6];
    const float* Wf1     = (const float*)d_in[7];
    const float* Wf2     = (const float*)d_in[8];
    const float* Wb      = (const float*)d_in[9];
    const float* A_log   = (const float*)d_in[10];
    const float* dt_bias = (const float*)d_in[11];
    const float* Wg      = (const float*)d_in[12];
    const float* bg      = (const float*)d_in[13];
    const float* norm_w  = (const float*)d_in[14];
    const float* Wo      = (const float*)d_in[15];
    float* out = (float*)d_out;

    const int M  = B_ * T_;              // 2048
    const size_t BIG = (size_t)M * D_;   // 2048*2048 floats

    float* ws   = (float*)d_ws;
    float* qp   = ws;                    // [M, 2048]
    float* kp   = qp   + BIG;
    float* vp   = kp   + BIG;
    float* gp   = vp   + BIG;            // hidden @ Wg
    float* f1   = gp   + BIG;            // [M, 128]
    float* g2   = f1   + (size_t)M * 128;
    float* bp   = g2   + (size_t)M * 128;     // [M, 16]
    float* beta = bp   + (size_t)M * H_;
    float* geff = beta + (size_t)M * H_;      // [M, 2048]
    float* q    = geff + BIG;
    float* k    = q    + BIG;
    float* v    = k    + BIG;
    float* o    = qp;                    // reuse (qp dead after conv)
    float* y    = kp;                    // reuse (kp dead after conv)

    dim3 blk(256);
    auto gemm = [&](const float* A, const float* Bm, float* C, int m, int n, int kd) {
        if ((n & 127) == 0) {
            dim3 grid(n / 128, m / 128);
            wmma_gemm_bf16_fast<<<grid, blk, 0, stream>>>(A, Bm, C, m, n, kd);
        } else {
            dim3 grid(n / 16, m / 128);
            wmma_gemm_bf16<<<grid, blk, 0, stream>>>(A, Bm, C, m, n, kd);
        }
    };

    // Projections (WMMA bf16)
    gemm(hidden, Wq,  qp, M, D_, D_);
    gemm(hidden, Wk,  kp, M, D_, D_);
    gemm(hidden, Wv,  vp, M, D_, D_);
    gemm(hidden, Wg,  gp, M, D_, D_);
    gemm(hidden, Wf1, f1, M, 128, D_);
    gemm(hidden, Wb,  bp, M, 16,  D_);
    gemm(f1,     Wf2, g2, M, 128, 128);

    // Conv + SiLU (+ L2 norm for q,k)
    dim3 cgrid(B_ * T_, H_);
    conv_silu_norm<<<cgrid, 128, 0, stream>>>(qp, conv_q, q, T_, 1, SCALE_);
    conv_silu_norm<<<cgrid, 128, 0, stream>>>(kp, conv_k, k, T_, 1, 1.0f);
    conv_silu_norm<<<cgrid, 128, 0, stream>>>(vp, conv_v, v, T_, 0, 1.0f);

    // Gates
    {
        int total = M * H_ * DK_;
        gate_kernel<<<(total + 255) / 256, 256, 0, stream>>>(g2, dt_bias, A_log, geff, total);
        int tb = M * H_;
        sigmoid_kernel<<<(tb + 255) / 256, 256, 0, stream>>>(bp, beta, tb);
    }

    // Delta-rule recurrence (state in VGPRs, one block per (b,h))
    recurrence_kernel<<<B_ * H_, 256, 0, stream>>>(q, k, v, geff, beta, o, T_);

    // Gated RMSNorm
    out_norm_kernel<<<M * H_, 128, 0, stream>>>(o, gp, bg, norm_w, y);

    // Output projection (WMMA bf16)
    gemm(y, Wo, out, M, D_, D_);
}
